// SoftFreezeAttention_18339510354292
// MI455X (gfx1250) — matrix operand order repaired and checked by reading, not compiled, not run on hardware
//
#include <hip/hip_runtime.h>

// ---------------------------------------------------------------------------
// SoftFreezeAttention for MI455X (gfx1250, wave32, WMMA 16x16x32 bf16)
//   M = 8*2048 = 16384 query rows, N = 4096 neurons, D = 512
//   out = softmax_rows(exp(-cdist(x,P)/tau)) @ V     (sum-normalized, not max)
// Fused flash-style: never materialize the 16384x4096 weight matrix.
// GEMM1 uses split-bf16 (hi+lo, 3 WMMAs) for ~fp32 distance accuracy.
// ---------------------------------------------------------------------------

#define EMBED   512
#define NNEUR   4096
#define MROWS   16384
#define MTILE   32
#define NCHUNK  64
#define NCHUNKS (NNEUR / NCHUNK)   // 64

typedef __attribute__((ext_vector_type(16))) __bf16 v16bf;
typedef __attribute__((ext_vector_type(8)))  float  v8f;

union ABFrag {
    v16bf v;
    uint4 u[2];
};

__device__ __forceinline__ unsigned short f2bf(float f) {
    unsigned int u = __float_as_uint(f);
    unsigned int r = u + 0x7FFFu + ((u >> 16) & 1u);   // round-to-nearest-even
    return (unsigned short)(r >> 16);
}
__device__ __forceinline__ float bf2f(unsigned short h) {
    return __uint_as_float(((unsigned int)h) << 16);
}

// ---------------------------------------------------------------------------
// Prep: split positions into bf16 hi/lo, exact fp32 row norms, 1/tau_eff.
// One 128-thread block per neuron.
// ---------------------------------------------------------------------------
__global__ void sfa_prep_neurons(const float* __restrict__ pos,
                                 const float* __restrict__ temp,
                                 const float* __restrict__ scale,
                                 unsigned short* __restrict__ p_hi,
                                 unsigned short* __restrict__ p_lo,
                                 float* __restrict__ p2,
                                 float* __restrict__ invt) {
    const int n = blockIdx.x;
    const int t = threadIdx.x;
    float s = 0.f;
    for (int d = t; d < EMBED; d += 128) {
        float v = pos[n * EMBED + d];
        s += v * v;
        unsigned short hi = f2bf(v);
        p_hi[n * EMBED + d] = hi;
        p_lo[n * EMBED + d] = f2bf(v - bf2f(hi));
    }
    __shared__ float red[128];
    red[t] = s;
    __syncthreads();
    for (int off = 64; off > 0; off >>= 1) {
        if (t < off) red[t] += red[t + off];
        __syncthreads();
    }
    if (t == 0) {
        p2[n]   = red[0];
        invt[n] = 1.0f / ((fabsf(temp[n]) + 0.1f) * scale[n]);
    }
}

// ---------------------------------------------------------------------------
// Prep: split x rows into bf16 hi/lo + exact fp32 row norms.
// One 128-thread block per row.
// ---------------------------------------------------------------------------
__global__ void sfa_prep_x(const float* __restrict__ x,
                           unsigned short* __restrict__ x_hi,
                           unsigned short* __restrict__ x_lo,
                           float* __restrict__ x2) {
    const int m = blockIdx.x;
    const int t = threadIdx.x;
    float s = 0.f;
    for (int d = t; d < EMBED; d += 128) {
        float v = x[m * EMBED + d];
        s += v * v;
        unsigned short hi = f2bf(v);
        x_hi[m * EMBED + d] = hi;
        x_lo[m * EMBED + d] = f2bf(v - bf2f(hi));
    }
    __shared__ float red[128];
    red[t] = s;
    __syncthreads();
    for (int off = 64; off > 0; off >>= 1) {
        if (t < off) red[t] += red[t + off];
        __syncthreads();
    }
    if (t == 0) x2[m] = red[0];
}

// ---------------------------------------------------------------------------
// Prep: V [N][D] fp32 -> V^T [D][N] bf16 (so GEMM2 B-fragments are contiguous)
// 64x64 tiles, 256 threads.
// ---------------------------------------------------------------------------
__global__ void sfa_prep_valuesT(const float* __restrict__ val,
                                 unsigned short* __restrict__ vT) {
    __shared__ unsigned short tile[64][72];  // padded vs bank conflicts
    const int n0 = blockIdx.x * 64;
    const int d0 = blockIdx.y * 64;
    const int t  = threadIdx.x;
    for (int e = t; e < 64 * 64; e += 256) {
        int i = e >> 6, j = e & 63;                       // i = n-local, j = d-local
        tile[i][j] = f2bf(val[(n0 + i) * EMBED + d0 + j]); // coalesced read
    }
    __syncthreads();
    for (int e = t; e < 64 * 64; e += 256) {
        int i = e >> 6, j = e & 63;                       // i = d-local, j = n-local
        vT[(d0 + i) * NNEUR + n0 + j] = tile[j][i];        // coalesced write
    }
}

// ---------------------------------------------------------------------------
// Main fused kernel. 256 threads = 8 waves per block; block owns 32 rows.
//   wave w: r = w&1 selects 16-row block; q = w>>1 in 0..3 selects
//     GEMM1 n-column subtile (q*16) and GEMM2 D-slab (q*128).
// ---------------------------------------------------------------------------
__global__ void __launch_bounds__(256) sfa_main(
    const unsigned short* __restrict__ xb_hi,   // [M][512] bf16
    const unsigned short* __restrict__ xb_lo,
    const unsigned short* __restrict__ pb_hi,   // [N][512] bf16
    const unsigned short* __restrict__ pb_lo,
    const unsigned short* __restrict__ vT,      // [512][N] bf16
    const float* __restrict__ x2,
    const float* __restrict__ p2,
    const float* __restrict__ invt,
    float* __restrict__ out)                    // [M][512] fp32
{
    __shared__ unsigned short x_hi_lds[MTILE][EMBED];   // 32 KB
    __shared__ unsigned short x_lo_lds[MTILE][EMBED];   // 32 KB
    __shared__ unsigned short w_lds[MTILE][NCHUNK];     // 4 KB
    __shared__ float rowsum[MTILE];

    const int tid   = threadIdx.x;
    const int wave  = tid >> 5;
    const int lane  = tid & 31;
    const int lhalf = lane >> 4;
    const int l16   = lane & 15;
    const int mbase = blockIdx.x * MTILE;

    // Stage the 32x512 bf16 x tile (hi+lo), 16B vector copies.
    for (int e8 = tid; e8 < MTILE * EMBED / 8; e8 += 256) {
        int r = e8 >> 6, c = (e8 & 63) * 8;
        *(uint4*)&x_hi_lds[r][c] = *(const uint4*)&xb_hi[(mbase + r) * EMBED + c];
        *(uint4*)&x_lo_lds[r][c] = *(const uint4*)&xb_lo[(mbase + r) * EMBED + c];
    }
    if (tid < MTILE) rowsum[tid] = 1e-8f;
    __syncthreads();

    const int r = wave & 1;      // 16-row block within tile
    const int q = wave >> 1;     // 0..3

    // x2 for this wave's rows (C/D layout: VGPR j -> row j + 8*lhalf)
    float x2r[8];
#pragma unroll
    for (int j = 0; j < 8; ++j)
        x2r[j] = x2[mbase + r * 16 + j + 8 * lhalf];

    v8f outacc[8];
#pragma unroll
    for (int s = 0; s < 8; ++s) outacc[s] = (v8f)(0.0f);
    float rs[8];
#pragma unroll
    for (int j = 0; j < 8; ++j) rs[j] = 0.0f;

    const int arow = r * 16 + l16;           // A-fragment row (lanes 0-15 / 16-31 same row)
    const int wrow = r * 16 + l16;           // GEMM2 A row in W tile

    for (int c = 0; c < NCHUNKS; ++c) {
        // ---------------- GEMM1: S = x @ P^T (split bf16, fp32 accum) -------
        v8f acc = (v8f)(0.0f);
        const int bn = c * NCHUNK + q * 16 + l16;   // this lane's neuron column
#pragma unroll
        for (int kk = 0; kk < 16; ++kk) {
            const int ka = kk * 32 + lhalf * 8;     // A: two contiguous 8-elem runs
            const int kb = kk * 32 + lhalf * 8;     // B: two contiguous 8-elem runs
            ABFrag ah, al, bh, bl;
            ah.u[0] = *(const uint4*)&x_hi_lds[arow][ka];
            ah.u[1] = *(const uint4*)&x_hi_lds[arow][ka + 16];
            al.u[0] = *(const uint4*)&x_lo_lds[arow][ka];
            al.u[1] = *(const uint4*)&x_lo_lds[arow][ka + 16];
            bh.u[0] = *(const uint4*)&pb_hi[bn * EMBED + kb];
            bh.u[1] = *(const uint4*)&pb_hi[bn * EMBED + kb + 16];
            bl.u[0] = *(const uint4*)&pb_lo[bn * EMBED + kb];
            bl.u[1] = *(const uint4*)&pb_lo[bn * EMBED + kb + 16];
            acc = __builtin_amdgcn_wmma_f32_16x16x32_bf16(false, al.v, false, bh.v,
                                                          (short)0, acc, false, false);
            acc = __builtin_amdgcn_wmma_f32_16x16x32_bf16(false, ah.v, false, bl.v,
                                                          (short)0, acc, false, false);
            acc = __builtin_amdgcn_wmma_f32_16x16x32_bf16(false, ah.v, false, bh.v,
                                                          (short)0, acc, false, false);
        }

        // ---------------- epilogue: d2 -> dist -> weight --------------------
        const float p2v = p2[bn];
        const float itv = invt[bn];
#pragma unroll
        for (int j = 0; j < 8; ++j) {
            float d2   = x2r[j] + p2v - 2.0f * acc[j];
            float dist = sqrtf(fmaxf(d2, 0.0f));
            float wgt  = __expf(-dist * itv);
            rs[j] += wgt;
            w_lds[r * 16 + j + 8 * lhalf][q * 16 + l16] = f2bf(wgt);
        }
        __syncthreads();   // W tile visible to all waves of this row block

        // ---------------- GEMM2: out += W @ V (bf16) ------------------------
#pragma unroll
        for (int sub = 0; sub < 8; ++sub) {
            const int dcol = q * 128 + sub * 16 + l16;
#pragma unroll
            for (int ks = 0; ks < 2; ++ks) {
                const int ka = ks * 32 + lhalf * 8;
                const int nb = c * NCHUNK + ks * 32 + lhalf * 8;
                ABFrag af, bf;
                af.u[0] = *(const uint4*)&w_lds[wrow][ka];
                af.u[1] = *(const uint4*)&w_lds[wrow][ka + 16];
                bf.u[0] = *(const uint4*)&vT[dcol * NNEUR + nb];
                bf.u[1] = *(const uint4*)&vT[dcol * NNEUR + nb + 16];
                outacc[sub] = __builtin_amdgcn_wmma_f32_16x16x32_bf16(
                    false, af.v, false, bf.v, (short)0, outacc[sub], false, false);
            }
        }
        __syncthreads();   // before next chunk overwrites W tile
    }

    // ---------------- row-sum reduction + normalized store ------------------
#pragma unroll
    for (int j = 0; j < 8; ++j)
        atomicAdd(&rowsum[r * 16 + j + 8 * lhalf], rs[j]);   // ds_add_f32
    __syncthreads();

    float rsum[8];
#pragma unroll
    for (int j = 0; j < 8; ++j)
        rsum[j] = rowsum[r * 16 + j + 8 * lhalf];

#pragma unroll
    for (int sub = 0; sub < 8; ++sub) {
        const int dcol = q * 128 + sub * 16 + l16;
#pragma unroll
        for (int j = 0; j < 8; ++j) {
            const int m = mbase + r * 16 + j + 8 * lhalf;
            out[m * EMBED + dcol] = outacc[sub][j] / rsum[j];
        }
    }
}

// ---------------------------------------------------------------------------
// Host launcher. Inputs: x, positions, values, temperature, neuron_scale.
// ---------------------------------------------------------------------------
extern "C" void kernel_launch(void* const* d_in, const int* in_sizes, int n_in,
                              void* d_out, int out_size, void* d_ws, size_t ws_size,
                              hipStream_t stream) {
    (void)in_sizes; (void)n_in; (void)out_size; (void)ws_size;

    const float* x     = (const float*)d_in[0];
    const float* pos   = (const float*)d_in[1];
    const float* val   = (const float*)d_in[2];
    const float* temp  = (const float*)d_in[3];
    const float* scale = (const float*)d_in[4];
    float* out = (float*)d_out;

    // Workspace layout (256B aligned slices)
    char* ws = (char*)d_ws;
    size_t off = 0;
    auto alloc = [&](size_t bytes) {
        char* p = ws + off;
        off += (bytes + 255) & ~(size_t)255;
        return p;
    };
    unsigned short* x_hi = (unsigned short*)alloc((size_t)MROWS * EMBED * 2);
    unsigned short* x_lo = (unsigned short*)alloc((size_t)MROWS * EMBED * 2);
    unsigned short* p_hi = (unsigned short*)alloc((size_t)NNEUR * EMBED * 2);
    unsigned short* p_lo = (unsigned short*)alloc((size_t)NNEUR * EMBED * 2);
    unsigned short* vT   = (unsigned short*)alloc((size_t)EMBED * NNEUR * 2);
    float* x2   = (float*)alloc((size_t)MROWS * 4);
    float* p2   = (float*)alloc((size_t)NNEUR * 4);
    float* invt = (float*)alloc((size_t)NNEUR * 4);

    sfa_prep_neurons<<<NNEUR, 128, 0, stream>>>(pos, temp, scale, p_hi, p_lo, p2, invt);
    sfa_prep_x<<<MROWS, 128, 0, stream>>>(x, x_hi, x_lo, x2);
    dim3 tgrid(NNEUR / 64, EMBED / 64);
    sfa_prep_valuesT<<<tgrid, 256, 0, stream>>>(val, vT);

    sfa_main<<<MROWS / MTILE, 256, 0, stream>>>(x_hi, x_lo, p_hi, p_lo, vT,
                                                x2, p2, invt, out);
}
